// OptimizedMoEAttention_31086973288490
// MI455X (gfx1250) — matrix operand order repaired and checked by reading, not compiled, not run on hardware
//
#include <hip/hip_runtime.h>
#include <hip/hip_bf16.h>
#include <stdint.h>

// Problem constants (match reference): B=4, S=512, D=1024, H=16, HD=64, E=4, K(top)=2
#define DEV __device__ __forceinline__

typedef __attribute__((ext_vector_type(16))) __bf16 v16bf;
typedef __attribute__((ext_vector_type(8)))  float  v8f;

union FragAB { v16bf v; uint4 q[2]; };
union FragC  { v8f  v; float f[8]; };

DEV unsigned short f2bf(float x) {           // RNE float -> bf16
  unsigned u = __float_as_uint(x);
  u += 0x7FFFu + ((u >> 16) & 1u);
  return (unsigned short)(u >> 16);
}
DEV float bf2f(unsigned short h) { return __uint_as_float(((unsigned)h) << 16); }

// ---------------------------------------------------------------------------
// Transpose + convert: in [K,N] f32  ->  out [N,K] bf16   (batched over z)
// ---------------------------------------------------------------------------
__global__ __launch_bounds__(256)
void transpose_f32_bf16(const float* __restrict__ in, unsigned short* __restrict__ out,
                        int K, int N) {
  __shared__ float tile[32][33];
  size_t base = (size_t)blockIdx.z * K * N;
  in += base; out += base;
  int n0 = blockIdx.x * 32, k0 = blockIdx.y * 32;
  int tx = threadIdx.x, ty = threadIdx.y;
  #pragma unroll
  for (int i = 0; i < 4; ++i)
    tile[ty + i * 8][tx] = in[(size_t)(k0 + ty + i * 8) * N + n0 + tx];
  __syncthreads();
  #pragma unroll
  for (int i = 0; i < 4; ++i)
    out[(size_t)(n0 + ty + i * 8) * K + k0 + tx] = f2bf(tile[tx][ty + i * 8]);
}

__global__ __launch_bounds__(256)
void f32_to_bf16(const float* __restrict__ in, unsigned short* __restrict__ out, int n) {
  for (int i = blockIdx.x * blockDim.x + threadIdx.x; i < n; i += gridDim.x * blockDim.x)
    out[i] = f2bf(in[i]);
}

// ---------------------------------------------------------------------------
// WMMA GEMM: C[M,N] = A[M,Kd](bf16) x Bt[N,Kd]^T(bf16) + bias, 128x128x32 tiles
// mode 0: out bf16 [M,N] = (c+bias)*scale
// mode 1: out bf16 scattered to Vt [B,H,HD,S]  (V projection, transposed store)
// mode 2: out f32  [M,N] (+)= emask[m,expert] * (c+bias)   (O-proj + combine)
// ---------------------------------------------------------------------------
__global__ __launch_bounds__(256)
void gemm_bf16_128x128(const unsigned short* __restrict__ A,
                       const unsigned short* __restrict__ Bt,
                       const float* __restrict__ bias,
                       void* __restrict__ outp,
                       int M, int N, int Kd, float scale, int mode,
                       const float* __restrict__ emask, int expert, int accflag) {
  __shared__ __align__(16) unsigned short As[128 * 40];
  __shared__ __align__(16) unsigned short Bs[128 * 40];
  const int m0 = blockIdx.x * 128, n0 = blockIdx.y * 128;
  const unsigned lane = threadIdx.x & 31u, wave = threadIdx.x >> 5;
  const unsigned lr = lane & 15u, lh = lane >> 4;        // row-in-tile, 16-lane half
  const unsigned wm = (wave >> 1) * 32u, wn = (wave & 1u) * 64u;

  FragC acc[2][4];
  #pragma unroll
  for (int i = 0; i < 2; ++i)
    #pragma unroll
    for (int j = 0; j < 4; ++j)
      #pragma unroll
      for (int r = 0; r < 8; ++r) acc[i][j].f[r] = 0.f;

  for (int k0 = 0; k0 < Kd; k0 += 32) {
    __syncthreads();
    #pragma unroll
    for (int i = 0; i < 2; ++i) {                        // 512 x 16B per tile
      int v = (int)threadIdx.x + i * 256;
      int row = v >> 2, kp = (v & 3) * 8;
      *(uint4*)&As[row * 40 + kp] = *(const uint4*)&A [(size_t)(m0 + row) * Kd + k0 + kp];
      *(uint4*)&Bs[row * 40 + kp] = *(const uint4*)&Bt[(size_t)(n0 + row) * Kd + k0 + kp];
    }
    __syncthreads();
    FragAB a[2], b[4];
    #pragma unroll
    for (int mf = 0; mf < 2; ++mf) {                     // A frag: K {kb..+7, kb+16..+23}
      int row = wm + mf * 16 + lr, kb = lh * 8;
      a[mf].q[0] = *(const uint4*)&As[row * 40 + kb];
      a[mf].q[1] = *(const uint4*)&As[row * 40 + kb + 16];
    }
    #pragma unroll
    for (int nf = 0; nf < 4; ++nf) {                     // B frag: K {kb..+7, kb+16..+23}
      int row = wn + nf * 16 + lr, ks = lh * 8;
      b[nf].q[0] = *(const uint4*)&Bs[row * 40 + ks];
      b[nf].q[1] = *(const uint4*)&Bs[row * 40 + ks + 16];
    }
    #pragma unroll
    for (int mf = 0; mf < 2; ++mf)
      #pragma unroll
      for (int nf = 0; nf < 4; ++nf)
        acc[mf][nf].v = __builtin_amdgcn_wmma_f32_16x16x32_bf16(
            false, a[mf].v, false, b[nf].v, (short)0, acc[mf][nf].v, false, false);
  }

  #pragma unroll
  for (int mf = 0; mf < 2; ++mf)
    #pragma unroll
    for (int nf = 0; nf < 4; ++nf) {
      int ng = n0 + wn + nf * 16 + lr;
      float bv = bias[ng];
      #pragma unroll
      for (int r = 0; r < 8; ++r) {
        int mg = m0 + wm + mf * 16 + r + lh * 8;
        float val = (acc[mf][nf].f[r] + bv) * scale;
        if (mode == 0) {
          ((unsigned short*)outp)[(size_t)mg * N + ng] = f2bf(val);
        } else if (mode == 1) {                          // Vt[b][h][hd][s]
          int h = ng >> 6, hd = ng & 63, bb = mg >> 9, s = mg & 511;
          ((unsigned short*)outp)[(((size_t)bb * 16 + h) * 64 + hd) * 512 + s] = f2bf(val);
        } else {
          float w = emask[(size_t)mg * 4 + expert];
          float* o = (float*)outp;
          size_t idx = (size_t)mg * N + ng;
          float res = w * val;
          if (accflag) o[idx] += res; else o[idx] = res;
        }
      }
    }
}

// ---------------------------------------------------------------------------
// Router tail: LayerNorm -> ReLU -> logits (D/2 x 4) -> softmax -> top-2 mask
// One 256-thread block per token; h row comes in as bf16 from the WMMA GEMM.
// ---------------------------------------------------------------------------
__global__ __launch_bounds__(256)
void router_finish(const unsigned short* __restrict__ hb,
                   const float* __restrict__ g, const float* __restrict__ bta,
                   const float* __restrict__ Wr2, const float* __restrict__ br2,
                   float* __restrict__ emask) {
  __shared__ float red[256];
  __shared__ float lg[4];
  int t = blockIdx.x, tid = threadIdx.x;
  float v0 = bf2f(hb[(size_t)t * 512 + tid * 2]);
  float v1 = bf2f(hb[(size_t)t * 512 + tid * 2 + 1]);

  red[tid] = v0 + v1;
  __syncthreads();
  for (int s = 128; s > 0; s >>= 1) { if (tid < s) red[tid] += red[tid + s]; __syncthreads(); }
  float mu = red[0] * (1.f / 512.f);
  __syncthreads();
  red[tid] = v0 * v0 + v1 * v1;
  __syncthreads();
  for (int s = 128; s > 0; s >>= 1) { if (tid < s) red[tid] += red[tid + s]; __syncthreads(); }
  float var = red[0] * (1.f / 512.f) - mu * mu;
  float rsq = rsqrtf(var + 1e-5f);
  float h0 = fmaxf((v0 - mu) * rsq * g[tid * 2]     + bta[tid * 2],     0.f);
  float h1 = fmaxf((v1 - mu) * rsq * g[tid * 2 + 1] + bta[tid * 2 + 1], 0.f);
  __syncthreads();
  for (int e = 0; e < 4; ++e) {
    red[tid] = h0 * Wr2[(tid * 2) * 4 + e] + h1 * Wr2[(tid * 2 + 1) * 4 + e];
    __syncthreads();
    for (int s = 128; s > 0; s >>= 1) { if (tid < s) red[tid] += red[tid + s]; __syncthreads(); }
    if (tid == 0) lg[e] = red[0] + br2[e];
    __syncthreads();
  }
  if (tid == 0) {
    float mx = fmaxf(fmaxf(lg[0], lg[1]), fmaxf(lg[2], lg[3]));
    float p[4], s = 0.f;
    for (int e = 0; e < 4; ++e) { p[e] = __expf(lg[e] - mx); s += p[e]; }
    float inv = 1.f / s;
    for (int e = 0; e < 4; ++e) p[e] *= inv;
    int i1 = 0; for (int e = 1; e < 4; ++e) if (p[e] > p[i1]) i1 = e;
    int i2 = -1; for (int e = 0; e < 4; ++e) { if (e == i1) continue; if (i2 < 0 || p[e] > p[i2]) i2 = e; }
    for (int e = 0; e < 4; ++e) emask[(size_t)t * 4 + e] = (e == i1 || e == i2) ? p[e] : 0.f;
  }
}

// ---------------------------------------------------------------------------
// Flash attention, one wave per 16-query tile per (b,h). Q pre-scaled 1/sqrt(HD).
// Q,K: bf16 [B*S, D] (head at h*64). Vt: bf16 [B,H,HD,S]. Out: bf16 [B*S, D].
// ---------------------------------------------------------------------------
__global__ __launch_bounds__(256)
void flash_attn(const unsigned short* __restrict__ Q,
                const unsigned short* __restrict__ Kb,
                const unsigned short* __restrict__ Vt,
                unsigned short* __restrict__ O) {
  __shared__ __align__(16) unsigned short pb[8][16][40];  // per-wave P repack buffer
  const unsigned lane = threadIdx.x & 31u, wave = threadIdx.x >> 5;
  const unsigned lr = lane & 15u, lh = lane >> 4;
  int bidx = blockIdx.x;               // B*H*4 blocks
  int b = bidx >> 6;
  int h = (bidx >> 2) & 15;
  int qblk = bidx & 3;
  int s0 = (qblk * 8 + (int)wave) * 16;

  // Q A-fragments for HD=64 (two K=32 steps), loaded once
  FragAB qa[2];
  const size_t rowQ = (size_t)(b * 512 + s0 + (int)lr) * 1024 + h * 64;
  #pragma unroll
  for (int kk = 0; kk < 2; ++kk) {
    int kb0 = kk * 32 + (int)lh * 8;
    qa[kk].q[0] = *(const uint4*)&Q[rowQ + kb0];
    qa[kk].q[1] = *(const uint4*)&Q[rowQ + kb0 + 16];
  }

  FragC acc[4];
  #pragma unroll
  for (int nf = 0; nf < 4; ++nf)
    #pragma unroll
    for (int r = 0; r < 8; ++r) acc[nf].f[r] = 0.f;
  float mrow[8], lrow[8];
  #pragma unroll
  for (int r = 0; r < 8; ++r) { mrow[r] = -1e30f; lrow[r] = 0.f; }

  for (int kt = 0; kt < 16; ++kt) {    // 32 keys per iteration
    // scores tile 16x32 = two 16x16 C frags
    FragC sf[2];
    #pragma unroll
    for (int g = 0; g < 2; ++g) {
      #pragma unroll
      for (int r = 0; r < 8; ++r) sf[g].f[r] = 0.f;
      #pragma unroll
      for (int kk = 0; kk < 2; ++kk) {
        FragAB kf;                     // B = K^T: lane = key, head-dim {kb..+7, kb+16..+23}
        int skey = kt * 32 + g * 16 + (int)lr;
        size_t rk = (size_t)(b * 512 + skey) * 1024 + h * 64 + kk * 32 + (int)lh * 8;
        kf.q[0] = *(const uint4*)&Kb[rk];
        kf.q[1] = *(const uint4*)&Kb[rk + 16];
        sf[g].v = __builtin_amdgcn_wmma_f32_16x16x32_bf16(
            false, qa[kk].v, false, kf.v, (short)0, sf[g].v, false, false);
      }
    }
    // online softmax update (row stats live per lane; reductions stay in 16-lane halves)
    float p0a[8], p1a[8];
    #pragma unroll
    for (int r = 0; r < 8; ++r) {
      float t = fmaxf(sf[0].f[r], sf[1].f[r]);
      #pragma unroll
      for (int d = 1; d < 16; d <<= 1) t = fmaxf(t, __shfl_xor(t, d, 32));
      float mnew = fmaxf(mrow[r], t);
      float c  = __expf(mrow[r] - mnew);
      float p0 = __expf(sf[0].f[r] - mnew);
      float p1 = __expf(sf[1].f[r] - mnew);
      float rs = p0 + p1;
      #pragma unroll
      for (int d = 1; d < 16; d <<= 1) rs += __shfl_xor(rs, d, 32);
      lrow[r] = lrow[r] * c + rs;
      mrow[r] = mnew;
      #pragma unroll
      for (int nf = 0; nf < 4; ++nf) acc[nf].f[r] *= c;
      p0a[r] = p0; p1a[r] = p1;
    }
    // repack P (C layout) -> A layout via per-wave LDS bounce
    #pragma unroll
    for (int r = 0; r < 8; ++r) {
      int row = r + (int)lh * 8;
      pb[wave][row][lr]      = f2bf(p0a[r]);
      pb[wave][row][16 + lr] = f2bf(p1a[r]);
    }
    asm volatile("s_wait_dscnt 0" ::: "memory");
    FragAB pf;
    { int kb0 = (int)lh * 8;
      pf.q[0] = *(const uint4*)&pb[wave][lr][kb0];
      pf.q[1] = *(const uint4*)&pb[wave][lr][kb0 + 16]; }
    // O += P x V (V transposed in memory: contiguous along keys)
    #pragma unroll
    for (int nf = 0; nf < 4; ++nf) {
      FragAB vf;
      int hd = nf * 16 + (int)lr;
      size_t rv = ((size_t)(b * 16 + h) * 64 + hd) * 512 + kt * 32 + (int)lh * 8;
      vf.q[0] = *(const uint4*)&Vt[rv];
      vf.q[1] = *(const uint4*)&Vt[rv + 16];
      acc[nf].v = __builtin_amdgcn_wmma_f32_16x16x32_bf16(
          false, pf.v, false, vf.v, (short)0, acc[nf].v, false, false);
    }
    asm volatile("" ::: "memory");
  }
  // finalize + store bf16
  float inv[8];
  #pragma unroll
  for (int r = 0; r < 8; ++r) inv[r] = 1.f / lrow[r];
  #pragma unroll
  for (int nf = 0; nf < 4; ++nf)
    #pragma unroll
    for (int r = 0; r < 8; ++r) {
      int s = s0 + r + (int)lh * 8;
      O[(size_t)(b * 512 + s) * 1024 + h * 64 + nf * 16 + lr] = f2bf(acc[nf].f[r] * inv[r]);
    }
}

// ---------------------------------------------------------------------------
extern "C" void kernel_launch(void* const* d_in, const int* in_sizes, int n_in,
                              void* d_out, int out_size, void* d_ws, size_t ws_size,
                              hipStream_t stream) {
  (void)in_sizes; (void)n_in; (void)out_size; (void)ws_size;
  const float* x   = (const float*)d_in[0];
  const float* Wq  = (const float*)d_in[1];
  const float* bq  = (const float*)d_in[2];
  const float* Wk  = (const float*)d_in[3];
  const float* bk  = (const float*)d_in[4];
  const float* Wv  = (const float*)d_in[5];
  const float* bv  = (const float*)d_in[6];
  const float* Wo  = (const float*)d_in[7];
  const float* bo  = (const float*)d_in[8];
  const float* Wr1 = (const float*)d_in[9];
  const float* br1 = (const float*)d_in[10];
  const float* lng = (const float*)d_in[11];
  const float* lnb = (const float*)d_in[12];
  const float* Wr2 = (const float*)d_in[13];
  const float* br2 = (const float*)d_in[14];

  char* ws = (char*)d_ws;
  size_t off = 0;
  auto alloc = [&](size_t bytes) -> char* {
    char* p = ws + off;
    off = (off + bytes + 255) & ~(size_t)255;
    return p;
  };
  unsigned short* xb   = (unsigned short*)alloc((size_t)2048 * 1024 * 2);
  unsigned short* wqT  = (unsigned short*)alloc((size_t)4 * 1024 * 1024 * 2);
  unsigned short* wkT  = (unsigned short*)alloc((size_t)4 * 1024 * 1024 * 2);
  unsigned short* wvT  = (unsigned short*)alloc((size_t)4 * 1024 * 1024 * 2);
  unsigned short* woT  = (unsigned short*)alloc((size_t)4 * 1024 * 1024 * 2);
  unsigned short* wr1T = (unsigned short*)alloc((size_t)512 * 1024 * 2);
  unsigned short* hb   = (unsigned short*)alloc((size_t)2048 * 512 * 2);
  float*          em   = (float*)         alloc((size_t)2048 * 4 * 4);
  unsigned short* qb   = (unsigned short*)alloc((size_t)2048 * 1024 * 2);
  unsigned short* kb   = (unsigned short*)alloc((size_t)2048 * 1024 * 2);
  unsigned short* vt   = (unsigned short*)alloc((size_t)4 * 16 * 64 * 512 * 2);
  unsigned short* ao   = (unsigned short*)alloc((size_t)2048 * 1024 * 2);

  dim3 b256(256), b32x8(32, 8);

  // Prep: weight transposes + x conversion
  transpose_f32_bf16<<<dim3(32, 32, 4), b32x8, 0, stream>>>(Wq, wqT, 1024, 1024);
  transpose_f32_bf16<<<dim3(32, 32, 4), b32x8, 0, stream>>>(Wk, wkT, 1024, 1024);
  transpose_f32_bf16<<<dim3(32, 32, 4), b32x8, 0, stream>>>(Wv, wvT, 1024, 1024);
  transpose_f32_bf16<<<dim3(32, 32, 4), b32x8, 0, stream>>>(Wo, woT, 1024, 1024);
  transpose_f32_bf16<<<dim3(16, 32, 1), b32x8, 0, stream>>>(Wr1, wr1T, 1024, 512);
  f32_to_bf16<<<dim3(1024), b256, 0, stream>>>(x, xb, 2048 * 1024);

  // Router: h = x @ Wr1 + br1 (WMMA), then LN/relu/logits/softmax/top-2
  gemm_bf16_128x128<<<dim3(16, 4), b256, 0, stream>>>(
      xb, wr1T, br1, hb, 2048, 512, 1024, 1.f, 0, nullptr, 0, 0);
  router_finish<<<dim3(2048), b256, 0, stream>>>(hb, lng, lnb, Wr2, br2, em);

  // Per-expert pipeline (sequential on stream)
  for (int e = 0; e < 4; ++e) {
    const unsigned short* wq_e = wqT + (size_t)e * 1024 * 1024;
    const unsigned short* wk_e = wkT + (size_t)e * 1024 * 1024;
    const unsigned short* wv_e = wvT + (size_t)e * 1024 * 1024;
    const unsigned short* wo_e = woT + (size_t)e * 1024 * 1024;
    gemm_bf16_128x128<<<dim3(16, 8), b256, 0, stream>>>(      // Q (pre-scaled 1/8)
        xb, wq_e, bq + e * 1024, qb, 2048, 1024, 1024, 0.125f, 0, nullptr, 0, 0);
    gemm_bf16_128x128<<<dim3(16, 8), b256, 0, stream>>>(      // K
        xb, wk_e, bk + e * 1024, kb, 2048, 1024, 1024, 1.f, 0, nullptr, 0, 0);
    gemm_bf16_128x128<<<dim3(16, 8), b256, 0, stream>>>(      // V -> transposed store
        xb, wv_e, bv + e * 1024, vt, 2048, 1024, 1024, 1.f, 1, nullptr, 0, 0);
    flash_attn<<<dim3(256), b256, 0, stream>>>(qb, kb, vt, ao);
    gemm_bf16_128x128<<<dim3(16, 8), b256, 0, stream>>>(      // O-proj + emask combine
        ao, wo_e, bo + e * 1024, d_out, 2048, 1024, 1024, 1.f, 2, em, e, e > 0 ? 1 : 0);
  }
}
